// Dual_Quantize_43645457662414
// MI455X (gfx1250) — matrix operand order repaired and checked by reading, not compiled, not run on hardware
//
#include <hip/hip_runtime.h>
#include <hip/hip_bf16.h>

typedef __bf16  v16bf __attribute__((ext_vector_type(16)));
typedef float   v8f   __attribute__((ext_vector_type(8)));

#define N_ROWS   65536
#define D_DIM    256
#define K_CODES  1024
#define TWO_D    512

// d_out layout (floats): q_hr, q_lr, diff_hr, diff_lr, ind, ind
#define Q_HR_OFF    0
#define Q_LR_OFF    (N_ROWS * D_DIM)
#define DIFF_HR_OFF (2 * N_ROWS * D_DIM)
#define DIFF_LR_OFF (DIFF_HR_OFF + 1)
#define IND0_OFF    (DIFF_HR_OFF + 2)
#define IND1_OFF    (IND0_OFF + N_ROWS)

// workspace layout (byte offsets)
#define WS_ET_HR   0                           // f32 [1024][256] transposed hr codebook
#define WS_ET_LR   (1024*256*4)                // f32 [1024][256] transposed lr codebook
#define WS_ET_BF   (2*1024*256*4)              // bf16 [1024][512] fused (lr||hr) transposed
#define WS_C2      (WS_ET_BF + 1024*512*2)     // f32 [1024] column norms
#define WS_IDX     (WS_C2 + 1024*4)            // i32 [65536] argmin indices

// ---------------------------------------------------------------------------
// Phase 0: transpose codebooks, build bf16 B-matrix, compute ||e||^2 per code
// ---------------------------------------------------------------------------
__global__ void __launch_bounds__(256) vq_prep(const float* __restrict__ embed_hr,
                                               const float* __restrict__ embed_lr,
                                               float* __restrict__ eT_hr,
                                               float* __restrict__ eT_lr,
                                               __bf16* __restrict__ eT_bf,
                                               float* __restrict__ c2,
                                               float* __restrict__ out) {
  const int k = blockIdx.x;   // code index 0..1023
  const int d = threadIdx.x;  // dim index  0..255
  const float vlr = embed_lr[(size_t)d * K_CODES + k];
  const float vhr = embed_hr[(size_t)d * K_CODES + k];
  eT_lr[(size_t)k * D_DIM + d] = vlr;
  eT_hr[(size_t)k * D_DIM + d] = vhr;
  eT_bf[(size_t)k * TWO_D + d]         = (__bf16)vlr;  // reduction dims 0..255   = lr
  eT_bf[(size_t)k * TWO_D + D_DIM + d] = (__bf16)vhr;  // reduction dims 256..511 = hr
  __shared__ float red[256];
  red[d] = vlr * vlr + vhr * vhr;
  __syncthreads();
  for (int s = 128; s > 0; s >>= 1) {
    if (d < s) red[d] += red[d + s];
    __syncthreads();
  }
  if (d == 0) c2[k] = red[0];
  if (blockIdx.x == 0 && d < 2) out[DIFF_HR_OFF + d] = 0.0f;  // zero MSE accumulators
}

// ---------------------------------------------------------------------------
// Phase 1: BF16 WMMA GEMM [N,512]x[512,K] fused with row-argmin over K.
// One wave = 16 rows; A fragments (full K-reduction) preloaded once.
// ---------------------------------------------------------------------------
__global__ void __launch_bounds__(256) vq_argmin(const float* __restrict__ in_hr,
                                                 const float* __restrict__ in_lr,
                                                 const __bf16* __restrict__ eT_bf,
                                                 const float* __restrict__ c2,
                                                 float* __restrict__ out,
                                                 int* __restrict__ ws_idx) {
  const int wave = threadIdx.x >> 5;
  const int lane = threadIdx.x & 31;
  const int m0   = (blockIdx.x * 8 + wave) * 16;   // this wave's 16-row tile
  const int mrow = m0 + (lane & 15);
  const bool hi  = lane >= 16;

  // Preload all 16 A fragments (K = 0..511). this code's contiguous k order (matches B):
  // lanes 0-15: K 0..15 (within chunk)
  // lanes16-31: K 16..31
  v16bf a[16];
#pragma unroll
  for (int c = 0; c < 16; ++c) {
    const int k0 = c * 32;
    const float* src = (k0 < D_DIM) ? in_lr : in_hr;   // flatten = concat(lr, hr)
    const float* p0 = src + (size_t)mrow * D_DIM + (k0 & (D_DIM - 1)) + (hi ? 16 : 0);
    const float* p1 = p0 + 8;
    v16bf f;
#pragma unroll
    for (int i = 0; i < 8; ++i) f[i] = (__bf16)p0[i];
#pragma unroll
    for (int i = 0; i < 8; ++i) f[8 + i] = (__bf16)p1[i];
    a[c] = f;
  }

  float minv[8];
  int   mini[8];
#pragma unroll
  for (int j = 0; j < 8; ++j) { minv[j] = __builtin_inff(); mini[j] = 0; }

  for (int n0 = 0; n0 < K_CODES; n0 += 16) {
    const int ncol = n0 + (lane & 15);
    // B in this code's contiguous k order (same as A): lanes 0-15 hold K 0..15, lanes 16-31 hold K 16..31
    const __bf16* brow = eT_bf + (size_t)ncol * TWO_D + (hi ? 16 : 0);
    v8f acc = {0.f, 0.f, 0.f, 0.f, 0.f, 0.f, 0.f, 0.f};
#pragma unroll
    for (int c = 0; c < 16; ++c) {
      v16bf b = *(const v16bf*)(brow + c * 32);
      acc = __builtin_amdgcn_wmma_f32_16x16x32_bf16(
          /*neg_a=*/false, a[c], /*neg_b=*/false, b,
          /*c_mod=*/(short)0, acc, /*reuse_a=*/false, /*reuse_b=*/false);
    }
    // dist(m,n) = ||f||^2 - 2 f.e + ||e||^2 ; ||f||^2 is row-constant -> drop
    const float cc = c2[ncol];
#pragma unroll
    for (int j = 0; j < 8; ++j) {
      const float s = __builtin_fmaf(-2.0f, acc[j], cc);
      if (s < minv[j]) { minv[j] = s; mini[j] = ncol; }
    }
  }

  // Cross-lane argmin over the 16 candidate columns held per half-wave.
  // C/D layout: lanes 0-15 VGPR j = row m0+j ; lanes 16-31 VGPR j = row m0+8+j.
#pragma unroll
  for (int off = 8; off >= 1; off >>= 1) {
#pragma unroll
    for (int j = 0; j < 8; ++j) {
      const float ov = __shfl_xor(minv[j], off, 32);
      const int   oi = __shfl_xor(mini[j], off, 32);
      if (ov < minv[j] || (ov == minv[j] && oi < mini[j])) { minv[j] = ov; mini[j] = oi; }
    }
  }

  if ((lane & 15) == 0) {
    const int rb = m0 + (hi ? 8 : 0);
#pragma unroll
    for (int j = 0; j < 8; ++j) {
      const int idx = mini[j];
      ws_idx[rb + j]          = idx;
      out[IND0_OFF + rb + j]  = (float)idx;
      out[IND1_OFF + rb + j]  = (float)idx;
    }
  }
}

// ---------------------------------------------------------------------------
// Phase 2: codebook gather (STE output == quantized vector) + MSE reduction.
// One wave per row (256 floats = 8 per lane, float4 vectorized).
// ---------------------------------------------------------------------------
__global__ void __launch_bounds__(256) vq_gather(const float* __restrict__ in_hr,
                                                 const float* __restrict__ in_lr,
                                                 const float* __restrict__ eT_hr,
                                                 const float* __restrict__ eT_lr,
                                                 const int* __restrict__ ws_idx,
                                                 float* __restrict__ out) {
  const int wave = threadIdx.x >> 5;
  const int lane = threadIdx.x & 31;
  const int row  = blockIdx.x * 8 + wave;
  const int idx  = ws_idx[row];

  const float4* qh = (const float4*)(eT_hr + (size_t)idx * D_DIM) + lane * 2;
  const float4* ql = (const float4*)(eT_lr + (size_t)idx * D_DIM) + lane * 2;
  const float4* ih = (const float4*)(in_hr + (size_t)row * D_DIM) + lane * 2;
  const float4* il = (const float4*)(in_lr + (size_t)row * D_DIM) + lane * 2;
  float4* oh = (float4*)(out + Q_HR_OFF + (size_t)row * D_DIM) + lane * 2;
  float4* ol = (float4*)(out + Q_LR_OFF + (size_t)row * D_DIM) + lane * 2;

  float sh = 0.f, sl = 0.f;
#pragma unroll
  for (int v = 0; v < 2; ++v) {
    float4 q = qh[v], x = ih[v];
    oh[v] = q;
    float dx = q.x - x.x, dy = q.y - x.y, dz = q.z - x.z, dw = q.w - x.w;
    sh += dx * dx + dy * dy + dz * dz + dw * dw;
    q = ql[v]; x = il[v];
    ol[v] = q;
    dx = q.x - x.x; dy = q.y - x.y; dz = q.z - x.z; dw = q.w - x.w;
    sl += dx * dx + dy * dy + dz * dz + dw * dw;
  }
#pragma unroll
  for (int off = 16; off >= 1; off >>= 1) {
    sh += __shfl_xor(sh, off, 32);
    sl += __shfl_xor(sl, off, 32);
  }
  if (lane == 0) {
    const float inv = 1.0f / (float)((size_t)N_ROWS * D_DIM);
    atomicAdd(&out[DIFF_HR_OFF], sh * inv);
    atomicAdd(&out[DIFF_LR_OFF], sl * inv);
  }
}

// ---------------------------------------------------------------------------
extern "C" void kernel_launch(void* const* d_in, const int* in_sizes, int n_in,
                              void* d_out, int out_size, void* d_ws, size_t ws_size,
                              hipStream_t stream) {
  const float* in_hr    = (const float*)d_in[0];
  const float* in_lr    = (const float*)d_in[1];
  const float* embed_hr = (const float*)d_in[2];
  const float* embed_lr = (const float*)d_in[3];
  float* out = (float*)d_out;

  char* ws = (char*)d_ws;
  float*  eT_hr  = (float*)(ws + WS_ET_HR);
  float*  eT_lr  = (float*)(ws + WS_ET_LR);
  __bf16* eT_bf  = (__bf16*)(ws + WS_ET_BF);
  float*  c2     = (float*)(ws + WS_C2);
  int*    ws_idx = (int*)(ws + WS_IDX);

  vq_prep<<<K_CODES, 256, 0, stream>>>(embed_hr, embed_lr, eT_hr, eT_lr, eT_bf, c2, out);
  vq_argmin<<<N_ROWS / 128, 256, 0, stream>>>(in_hr, in_lr, eT_bf, c2, out, ws_idx);
  vq_gather<<<N_ROWS / 8, 256, 0, stream>>>(in_hr, in_lr, eT_hr, eT_lr, ws_idx, out);
}
